// LigerFusedLinearJSD_6863357739727
// MI455X (gfx1250) — matrix operand order repaired and checked by reading, not compiled, not run on hardware
//
#include <hip/hip_runtime.h>
#include <hip/hip_bf16.h>
#include <math.h>

// Problem constants (from reference)
#define BT    2048
#define HS    1024
#define HT    2048
#define VOC   32000
#define IGNORE_IDX (-100)

// Tiling
#define BM      16                // token rows per block (one WMMA M-tile)
#define NWAVE   8                 // 256 threads = 8 wave32
#define NRBLK   (BT / BM)         // 128 row blocks
#define VSPLIT  4                 // vocab split -> 512 blocks for GEMM kernels
#define VCHUNK  (VOC / VSPLIT)    // 8000 vocab entries per chunk
#define NPAIRC  (VCHUNK / 32)     // 250 pairs of 16-col tiles per chunk
#define GRID    (NRBLK * VSPLIT)  // 512

typedef __attribute__((ext_vector_type(16))) __bf16 v16bf;
typedef __attribute__((ext_vector_type(8)))  float  v8f;

union BF16x16 { uint4 u[2]; unsigned short s[16]; v16bf v; };

__device__ inline unsigned short f2bf(float f) {
  unsigned int u = __float_as_uint(f);
  u += 0x7FFFu + ((u >> 16) & 1u);   // round to nearest even
  return (unsigned short)(u >> 16);
}

// ---------------- weight conversion: fp32 -> bf16 in workspace ----------------
__global__ void convert_w_kernel(const float4* __restrict__ src,
                                 uint2* __restrict__ dst, int n4) {
  int i = blockIdx.x * blockDim.x + threadIdx.x;
  int stride = blockDim.x * gridDim.x;
  for (; i < n4; i += stride) {
    float4 f = src[i];
    uint2 o;
    o.x = (unsigned int)f2bf(f.x) | ((unsigned int)f2bf(f.y) << 16);
    o.y = (unsigned int)f2bf(f.z) | ((unsigned int)f2bf(f.w) << 16);
    dst[i] = o;
  }
}

// ---------------- fragment loaders ----------------
// A: 16x32 bf16, row-major in LDS (row stride = H halves).
// lanes 0-15 -> M=lane, K = k0+{0..15}; lanes 16-31 -> K = k0+{16..31}.
template <int H>
__device__ inline v16bf load_a_frag(const unsigned short* lds, int k0, int lane) {
  const int row = lane & 15;
  const int kh  = (lane >> 4) << 4;       // 0 or 16
  const unsigned short* p = lds + row * H + k0 + kh;
  BF16x16 r;
  r.u[0] = *(const uint4*)(p);            // K kh .. kh+7
  r.u[1] = *(const uint4*)(p + 8);        // K kh+8 .. kh+15
  return r.v;
}

// B: 32x16 bf16. Lane = column N=lane&15, K-half (lane>>4)*16..+15:
// 16 contiguous elements of weight row (v0+N) starting at k0+(lane>>4)*16.
template <int H, bool BF16W>
__device__ inline v16bf load_b_frag(const void* W, int v0, int k0, int lane) {
  const int n  = lane & 15;
  const int ks = (lane >> 4) << 4;        // 0 or 16
  BF16x16 r;
  if constexpr (BF16W) {
    const unsigned short* p = (const unsigned short*)W + (size_t)(v0 + n) * H + (k0 + ks);
    r.u[0] = *(const uint4*)(p);
    r.u[1] = *(const uint4*)(p + 8);
  } else {
    const float* p = (const float*)W + (size_t)(v0 + n) * H + (k0 + ks);
#pragma unroll
    for (int i = 0; i < 16; i += 4) {
      float4 f = *(const float4*)(p + i);
      r.s[i + 0] = f2bf(f.x); r.s[i + 1] = f2bf(f.y);
      r.s[i + 2] = f2bf(f.z); r.s[i + 3] = f2bf(f.w);
    }
  }
  return r.v;
}

// 16-row x 32-col logit pair, K sweep with WMMA bf16 + f32 accumulate.
// Software-pipelined: prefetch next A/B fragments before issuing current WMMAs,
// so loads for step k+1 are in flight while step k computes.
template <int H, bool BF16W>
__device__ inline void gemm_pair(const unsigned short* Alds,
                                 const void* W, int v0, int lane,
                                 v8f& c0, v8f& c1) {
  v16bf a  = load_a_frag<H>(Alds, 0, lane);
  v16bf b0 = load_b_frag<H, BF16W>(W, v0,      0, lane);
  v16bf b1 = load_b_frag<H, BF16W>(W, v0 + 16, 0, lane);
#pragma unroll 4
  for (int k0 = 0; k0 < H; k0 += 32) {
    const int kn = k0 + 32;
    v16bf an = a, bn0 = b0, bn1 = b1;
    if (kn < H) {               // compile-time folded (H is a template constant)
      an  = load_a_frag<H>(Alds, kn, lane);
      bn0 = load_b_frag<H, BF16W>(W, v0,      kn, lane);
      bn1 = load_b_frag<H, BF16W>(W, v0 + 16, kn, lane);
    }
    c0 = __builtin_amdgcn_wmma_f32_16x16x32_bf16(false, a, false, b0, (short)0, c0, false, false);
    c1 = __builtin_amdgcn_wmma_f32_16x16x32_bf16(false, a, false, b1, (short)0, c1, false, false);
    a = an; b0 = bn0; b1 = bn1;
  }
}

// Stage the block's 16 activation rows into LDS as bf16 (once per kernel).
__device__ inline void stage_acts(const float* __restrict__ xs,
                                  const float* __restrict__ xt,
                                  unsigned short* As, unsigned short* At,
                                  int row0, int tid) {
  for (int i = tid; i < BM * HS; i += 256)
    As[i] = f2bf(xs[(size_t)(row0 + (i >> 10)) * HS + (i & (HS - 1))]);
  for (int i = tid; i < BM * HT; i += 256)
    At[i] = f2bf(xt[(size_t)(row0 + (i >> 11)) * HT + (i & (HT - 1))]);
}

__device__ inline float jsd_term(float lq, float lp) {
  // beta = 0.5
  float q = __expf(lq), p = __expf(lp);
  float m = 0.5f * (p + q);
  if (m <= 0.f) return 0.f;
  float lm = __logf(m);
  return 0.5f * (p * (lp - lm) + q * (lq - lm));
}

// ---------------- pass 1: partial sum(exp(z)) per row per vocab chunk ----------------
// Logit magnitudes here are tiny (|z| <~ 8), so direct expsum is safe in fp32.
template <bool BF16W>
__global__ __launch_bounds__(256)
void pass1_kernel(const float* __restrict__ xs, const float* __restrict__ xt,
                  const void* __restrict__ Wsv, const void* __restrict__ Wtv,
                  float2* __restrict__ stats /* [BT][VSPLIT] */) {
  __shared__ __align__(16) unsigned short As[BM * HS];
  __shared__ __align__(16) unsigned short At[BM * HT];
  __shared__ float red[NWAVE * 16 * 2];

  const int tid   = threadIdx.x;
  const int wave  = tid >> 5;
  const int lane  = tid & 31;
  const int row0  = blockIdx.x * BM;
  const int vb    = blockIdx.y * VCHUNK;
  const int rbase = (lane >> 4) << 3;   // C layout: lanes 0-15 rows 0..7, 16-31 rows 8..15

  stage_acts(xs, xt, As, At, row0, tid);
  __syncthreads();

  float sS[8], sT[8];
#pragma unroll
  for (int r = 0; r < 8; r++) { sS[r] = 0.f; sT[r] = 0.f; }

  for (int p = wave; p < NPAIRC; p += NWAVE) {
    const int v0 = vb + (p << 5);
    v8f zs0 = {0.f,0.f,0.f,0.f,0.f,0.f,0.f,0.f};
    v8f zs1 = zs0, zt0 = zs0, zt1 = zs0;
    gemm_pair<HS, BF16W>(As, Wsv, v0, lane, zs0, zs1);
    gemm_pair<HT, BF16W>(At, Wtv, v0, lane, zt0, zt1);
#pragma unroll
    for (int r = 0; r < 8; r++) {
      sS[r] += __expf(zs0[r]) + __expf(zs1[r]);
      sT[r] += __expf(zt0[r]) + __expf(zt1[r]);
    }
  }
  // sum across the 16 lanes sharing each row (xor masks stay inside halves)
#pragma unroll
  for (int off = 1; off < 16; off <<= 1) {
#pragma unroll
    for (int r = 0; r < 8; r++) {
      sS[r] += __shfl_xor(sS[r], off, 32);
      sT[r] += __shfl_xor(sT[r], off, 32);
    }
  }
  if ((lane & 15) == 0) {
#pragma unroll
    for (int r = 0; r < 8; r++) {
      red[(wave * 16 + rbase + r) * 2 + 0] = sS[r];
      red[(wave * 16 + rbase + r) * 2 + 1] = sT[r];
    }
  }
  __syncthreads();
  if (tid < 16) {
    float ss = 0.f, st = 0.f;
    for (int w = 0; w < NWAVE; w++) {
      ss += red[(w * 16 + tid) * 2 + 0];
      st += red[(w * 16 + tid) * 2 + 1];
    }
    stats[(size_t)(row0 + tid) * VSPLIT + blockIdx.y] = make_float2(ss, st);
  }
}

// ---------------- merge chunk expsums -> per-row logsumexp ----------------
__global__ void lse_kernel(const float2* __restrict__ stats,
                           float2* __restrict__ lse) {
  int i = blockIdx.x * blockDim.x + threadIdx.x;
  if (i < BT) {
    float ss = 0.f, st = 0.f;
    for (int c = 0; c < VSPLIT; c++) {
      float2 v = stats[(size_t)i * VSPLIT + c];
      ss += v.x; st += v.y;
    }
    lse[i] = make_float2(__logf(ss), __logf(st));
  }
}

// ---------------- pass 2: recompute logits, accumulate generalized JSD ----------------
template <bool BF16W>
__global__ __launch_bounds__(256)
void pass2_kernel(const float* __restrict__ xs, const float* __restrict__ xt,
                  const void* __restrict__ Wsv, const void* __restrict__ Wtv,
                  const float2* __restrict__ lse, const int* __restrict__ label,
                  float* __restrict__ partials /* [GRID] */) {
  __shared__ __align__(16) unsigned short As[BM * HS];
  __shared__ __align__(16) unsigned short At[BM * HT];
  __shared__ float red[NWAVE * 16];
  __shared__ float lS[16], lT[16];
  __shared__ float blocksum[16];

  const int tid   = threadIdx.x;
  const int wave  = tid >> 5;
  const int lane  = tid & 31;
  const int row0  = blockIdx.x * BM;
  const int vb    = blockIdx.y * VCHUNK;
  const int rbase = (lane >> 4) << 3;

  if (tid < 16) {
    float2 v = lse[row0 + tid];
    lS[tid] = v.x; lT[tid] = v.y;
  }
  stage_acts(xs, xt, As, At, row0, tid);
  __syncthreads();

  float jacc[8];
#pragma unroll
  for (int r = 0; r < 8; r++) jacc[r] = 0.f;

  for (int p = wave; p < NPAIRC; p += NWAVE) {
    const int v0 = vb + (p << 5);
    v8f zs0 = {0.f,0.f,0.f,0.f,0.f,0.f,0.f,0.f};
    v8f zs1 = zs0, zt0 = zs0, zt1 = zs0;
    gemm_pair<HS, BF16W>(As, Wsv, v0, lane, zs0, zs1);
    gemm_pair<HT, BF16W>(At, Wtv, v0, lane, zt0, zt1);
#pragma unroll
    for (int r = 0; r < 8; r++) {
      const float ls = lS[rbase + r];
      const float lt = lT[rbase + r];
      jacc[r] += jsd_term(zs0[r] - ls, zt0[r] - lt);
      jacc[r] += jsd_term(zs1[r] - ls, zt1[r] - lt);
    }
  }
#pragma unroll
  for (int off = 1; off < 16; off <<= 1) {
#pragma unroll
    for (int r = 0; r < 8; r++) jacc[r] += __shfl_xor(jacc[r], off, 32);
  }
  if ((lane & 15) == 0) {
#pragma unroll
    for (int r = 0; r < 8; r++) red[wave * 16 + rbase + r] = jacc[r];
  }
  __syncthreads();
  if (tid < 16) {
    float s = 0.f;
    for (int w = 0; w < NWAVE; w++) s += red[w * 16 + tid];
    blocksum[tid] = (label[row0 + tid] != IGNORE_IDX) ? s : 0.f;
  }
  __syncthreads();
  if (tid == 0) {
    float s = 0.f;
#pragma unroll
    for (int i = 0; i < 16; i++) s += blocksum[i];
    partials[blockIdx.y * NRBLK + blockIdx.x] = s;   // plain store -> deterministic
  }
}

// ---------------- deterministic finalize ----------------
__global__ void finalize_kernel(const float* __restrict__ partials, int nparts,
                                const int* __restrict__ label, int ntok,
                                float* __restrict__ out) {
  __shared__ float ssum[256];
  __shared__ int   scnt[256];
  const int tid = threadIdx.x;
  float s = 0.f; int c = 0;
  for (int i = tid; i < nparts; i += 256) s += partials[i];
  for (int i = tid; i < ntok;   i += 256) c += (label[i] != IGNORE_IDX) ? 1 : 0;
  ssum[tid] = s; scnt[tid] = c;
  __syncthreads();
  for (int off = 128; off > 0; off >>= 1) {
    if (tid < off) { ssum[tid] += ssum[tid + off]; scnt[tid] += scnt[tid + off]; }
    __syncthreads();
  }
  if (tid == 0) {
    int n = scnt[0] > 1 ? scnt[0] : 1;
    out[0] = ssum[0] / (float)n;
  }
}

extern "C" void kernel_launch(void* const* d_in, const int* in_sizes, int n_in,
                              void* d_out, int out_size, void* d_ws, size_t ws_size,
                              hipStream_t stream) {
  (void)in_sizes; (void)n_in; (void)out_size;
  const float* xs  = (const float*)d_in[0];   // (BT, HS)
  const float* Wsf = (const float*)d_in[1];   // (VOC, HS)
  const float* xt  = (const float*)d_in[2];   // (BT, HT)
  const float* Wtf = (const float*)d_in[3];   // (VOC, HT)
  const int*   lab = (const int*)d_in[4];     // (BT,)
  float*       out = (float*)d_out;

  const size_t nWs = (size_t)VOC * HS;
  const size_t nWt = (size_t)VOC * HT;
  const size_t bf16_bytes  = (nWs + nWt) * sizeof(unsigned short);
  const size_t stats_bytes = (size_t)BT * VSPLIT * sizeof(float2);
  const size_t lse_bytes   = (size_t)BT * sizeof(float2);
  const size_t part_bytes  = (size_t)GRID * sizeof(float);

  const dim3 gemm_grid(NRBLK, VSPLIT);

  if (ws_size >= bf16_bytes + stats_bytes + lse_bytes + part_bytes) {
    // Preferred path: pre-convert weights to bf16 (196 MB, ~L2 resident),
    // then the WMMA kernels stream bf16 weights.
    unsigned short* Wsb = (unsigned short*)d_ws;
    unsigned short* Wtb = Wsb + nWs;
    char* p = (char*)d_ws + bf16_bytes;
    float2* stats = (float2*)p;              p += stats_bytes;
    float2* lse   = (float2*)p;              p += lse_bytes;
    float*  parts = (float*)p;

    convert_w_kernel<<<2048, 256, 0, stream>>>((const float4*)Wsf, (uint2*)Wsb, (int)(nWs / 4));
    convert_w_kernel<<<2048, 256, 0, stream>>>((const float4*)Wtf, (uint2*)Wtb, (int)(nWt / 4));
    pass1_kernel<true><<<gemm_grid, 256, 0, stream>>>(xs, xt, Wsb, Wtb, stats);
    lse_kernel<<<(BT + 255) / 256, 256, 0, stream>>>(stats, lse);
    pass2_kernel<true><<<gemm_grid, 256, 0, stream>>>(xs, xt, Wsb, Wtb, lse, lab, parts);
    finalize_kernel<<<1, 256, 0, stream>>>(parts, GRID, lab, BT, out);
  } else {
    // Fallback: convert weights in-register at load time (reads fp32 directly).
    char* p = (char*)d_ws;
    float2* stats = (float2*)p;              p += stats_bytes;
    float2* lse   = (float2*)p;              p += lse_bytes;
    float*  parts = (float*)p;

    pass1_kernel<false><<<gemm_grid, 256, 0, stream>>>(xs, xt, Wsf, Wtf, stats);
    lse_kernel<<<(BT + 255) / 256, 256, 0, stream>>>(stats, lse);
    pass2_kernel<false><<<gemm_grid, 256, 0, stream>>>(xs, xt, Wsf, Wtf, lse, lab, parts);
    finalize_kernel<<<1, 256, 0, stream>>>(parts, GRID, lab, BT, out);
  }
}
